// HybridQuantumClassicalDetector_51565377355851
// MI455X (gfx1250) — matrix operand order repaired and checked by reading, not compiled, not run on hardware
//
#include <hip/hip_runtime.h>
#include <hip/hip_bf16.h>
#include <math.h>

#define BATCH    4096
#define IN_DIM   6
#define HID      16
#define NQ       12
#define NLAYERS  3
#define NSTATE   (1 << NQ)     // 4096 amplitudes
#define QTHREADS 256
#define PAIRS    (NSTATE / 2)  // 2048

typedef __attribute__((ext_vector_type(16))) _Float16 v16h;
typedef __attribute__((ext_vector_type(8)))  float    v8f;

// ---------------------------------------------------------------------------
// WMMA MLP tile engine: one wave32 computes a 16-sample tile through a layer.
// A fragment (16x32 f16): lane L -> M = L&15, Kbase = (L>>4)*8
//   VGPR v<4 : K = Kbase + 2v, 2v+1 ; VGPR v>=4 : K = Kbase + 16 + 2(v-4)
// B fragment (32x16 f16): lane L -> N = L&15, Kbase = (L>>4)*8, K order as A (h<8: Kbase+h, else Kbase+8+h)
// D fragment (16x16 f32): lane L -> N = L&15, VGPR r -> M = r + 8*(L>>4)
// A and B are staged in zero-padded LDS tiles so fragment building is
// branch-free (padding supplies the K/N masking).
// ---------------------------------------------------------------------------

__device__ __forceinline__ v8f tile_matmul(const float (*As)[32], const float (*Bs)[32],
                                           int M, int N, int kb8, int kb16)
{
    v16h a, b;
#pragma unroll
    for (int v = 0; v < 8; ++v) {
        int K0 = (v < 4) ? (kb8 + 2 * v) : (kb8 + 16 + 2 * (v - 4));
        a[2 * v]     = (_Float16)As[M][K0];
        a[2 * v + 1] = (_Float16)As[M][K0 + 1];
    }
#pragma unroll
    for (int h = 0; h < 16; ++h) {
        b[h] = (_Float16)Bs[N][(h < 8) ? (kb8 + h) : (kb8 + 8 + h)];
    }
    v8f c = {};
    return __builtin_amdgcn_wmma_f32_16x16x32_f16(false, a, false, b, (short)0, c,
                                                  false, false);
}

__global__ __launch_bounds__(32) void preproc_wmma(
    const float* __restrict__ x,
    const float* __restrict__ W1, const float* __restrict__ b1,
    const float* __restrict__ W2, const float* __restrict__ b2,
    const float* __restrict__ W3, const float* __restrict__ b3,
    float* __restrict__ angles)
{
    __shared__ float As[16][32];
    __shared__ float Bs[16][32];
    const int lane = threadIdx.x;
    const int r0   = blockIdx.x * 16;
    const int M    = lane & 15;
    const int half = lane >> 4;
    const int kb8  = half * 8;
    const int N    = lane & 15;
    const int kb16 = half * 16;

    __builtin_prefetch(W2, 0, 0);
    __builtin_prefetch(W3, 0, 0);

    // zero staging (padding provides K/N masking)
    float* Af = &As[0][0];
    float* Bf = &Bs[0][0];
    for (int i = lane; i < 16 * 32; i += 32) { Af[i] = 0.f; Bf[i] = 0.f; }
    __syncthreads();

    // stage x tile (16x6) and W1 (16x6), coalesced and branch-free
    for (int i = lane; i < 16 * IN_DIM; i += 32) {
        As[i / IN_DIM][i % IN_DIM] = x[(size_t)(r0 + i / IN_DIM) * IN_DIM + i % IN_DIM];
        Bs[i / IN_DIM][i % IN_DIM] = W1[i];
    }
    __syncthreads();

    // ---- Layer 1: h1 = relu(x @ W1^T + b1) ----
    v8f acc = tile_matmul(As, Bs, M, N, kb8, kb16);
    float bias = b1[N];
#pragma unroll
    for (int r = 0; r < 8; ++r) As[r + 8 * half][N] = fmaxf(acc[r] + bias, 0.f);
    // stage W2 (16x16); cols 16..31 stay zero
    for (int i = lane; i < 16 * 16; i += 32) Bs[i >> 4][i & 15] = W2[i];
    __syncthreads();

    // ---- Layer 2: h2 = relu(h1 @ W2^T + b2) ----
    acc  = tile_matmul(As, Bs, M, N, kb8, kb16);
    bias = b2[N];
#pragma unroll
    for (int r = 0; r < 8; ++r) As[r + 8 * half][N] = fmaxf(acc[r] + bias, 0.f);
    // stage W3 (12x16); zero rows 12..15 (stale W2 data)
    for (int i = lane; i < NQ * 16; i += 32) Bs[i >> 4][i & 15] = W3[i];
    for (int i = lane; i < 4 * 16; i += 32) Bs[12 + (i >> 4)][i & 15] = 0.f;
    __syncthreads();

    // ---- Layer 3: angles = h2 @ W3^T + b3 (N valid < 12) ----
    acc = tile_matmul(As, Bs, M, N, kb8, kb16);
    if (N < NQ) {
        bias = b3[N];
#pragma unroll
        for (int r = 0; r < 8; ++r) {
            int m = r + 8 * half;
            angles[(size_t)(r0 + m) * NQ + N] = acc[r] + bias;
        }
    }
}

// ---------------------------------------------------------------------------
// 12-qubit state-vector simulator: one workgroup per batch sample,
// full state (re/im) resident in LDS (32 KB).
// ---------------------------------------------------------------------------

__device__ __forceinline__ void apply1q(float* sr, float* si, int q,
    float u00r, float u00i, float u01r, float u01i,
    float u10r, float u10i, float u11r, float u11i, int tid)
{
    const unsigned mask = 1u << q;
    const unsigned low  = mask - 1u;
    for (int p = tid; p < PAIRS; p += QTHREADS) {
        unsigned i0 = (((unsigned)p & ~low) << 1) | ((unsigned)p & low);
        unsigned i1 = i0 | mask;
        float a0r = sr[i0], a0i = si[i0];
        float a1r = sr[i1], a1i = si[i1];
        sr[i0] = u00r * a0r - u00i * a0i + u01r * a1r - u01i * a1i;
        si[i0] = u00r * a0i + u00i * a0r + u01r * a1i + u01i * a1r;
        sr[i1] = u10r * a0r - u10i * a0i + u11r * a1r - u11i * a1i;
        si[i1] = u10r * a0i + u10i * a0r + u11r * a1i + u11i * a1r;
    }
    __syncthreads();
}

__device__ __forceinline__ void cnot_gate(float* sr, float* si, int cq, int tq, int tid)
{
    const unsigned mc = 1u << cq, mt = 1u << tq;
    const unsigned bl = (cq < tq) ? mc : mt;  // lower bit mask
    const unsigned bh = (cq < tq) ? mt : mc;  // higher bit mask
    const int npair = NSTATE / 4;             // 1024 swaps
    for (int p = tid; p < npair; p += QTHREADS) {
        unsigned xx = (unsigned)p;
        unsigned l1 = xx & (bl - 1u);
        xx = ((xx ^ l1) << 1) | l1;           // insert 0 at lower position
        unsigned l2 = xx & (bh - 1u);
        xx = ((xx ^ l2) << 1) | l2;           // insert 0 at higher position
        unsigned i0 = xx | mc;                // control=1, target=0
        unsigned i1 = xx | mc | mt;           // control=1, target=1
        float tr = sr[i0]; sr[i0] = sr[i1]; sr[i1] = tr;
        float ti = si[i0]; si[i0] = si[i1]; si[i1] = ti;
    }
    __syncthreads();
}

__global__ __launch_bounds__(QTHREADS) void quantum_sim(
    const float* __restrict__ angles,   // (BATCH, NQ)
    const float* __restrict__ qparams,  // (NLAYERS, NQ, 3): rx, ry, rz
    float* __restrict__ qexp)           // (BATCH, NQ)
{
    __shared__ float sr[NSTATE];
    __shared__ float si[NSTATE];
    __shared__ float red[QTHREADS / 32][NQ];
    const int tid = threadIdx.x;
    const int bs  = blockIdx.x;

    for (int k = tid; k < NSTATE; k += QTHREADS) {
        sr[k] = (k == 0) ? 1.f : 0.f;
        si[k] = 0.f;
    }
    __syncthreads();

    // angle-encoding RY gates (real 2x2)
    for (int qb = 0; qb < NQ; ++qb) {
        float th = angles[(size_t)bs * NQ + qb];
        float s, c;
        __sincosf(0.5f * th, &s, &c);
        apply1q(sr, si, qb, c, 0.f, -s, 0.f, s, 0.f, c, 0.f, tid);
    }

    for (int l = 0; l < NLAYERS; ++l) {
        for (int qb = 0; qb < NQ; ++qb) {
            const float* pp = qparams + ((size_t)l * NQ + qb) * 3;
            float sa, ca, sb, cb, sg, cg;
            __sincosf(0.5f * pp[0], &sa, &ca);   // RX
            __sincosf(0.5f * pp[1], &sb, &cb);   // RY
            __sincosf(0.5f * pp[2], &sg, &cg);   // RZ
            // M = Ry @ Rx
            float m00r =  cb * ca, m00i =  sb * sa;
            float m01r = -sb * ca, m01i = -cb * sa;
            float m10r =  sb * ca, m10i = -cb * sa;
            float m11r =  cb * ca, m11i = -sb * sa;
            // U = Rz @ M : row0 *= (cg, -sg); row1 *= (cg, +sg)
            float u00r = m00r * cg + m00i * sg, u00i = m00i * cg - m00r * sg;
            float u01r = m01r * cg + m01i * sg, u01i = m01i * cg - m01r * sg;
            float u10r = m10r * cg - m10i * sg, u10i = m10i * cg + m10r * sg;
            float u11r = m11r * cg - m11i * sg, u11i = m11i * cg + m11r * sg;
            apply1q(sr, si, qb, u00r, u00i, u01r, u01i, u10r, u10i, u11r, u11i, tid);
        }
        for (int i = 0; i < NQ - 1; ++i) cnot_gate(sr, si, i, i + 1, tid);
        cnot_gate(sr, si, NQ - 1, 0, tid);
    }

    // PauliZ expectations: ev[q] = sum p[idx] * (1 - 2*bit_q(idx))
    float acc[NQ];
#pragma unroll
    for (int q = 0; q < NQ; ++q) acc[q] = 0.f;
    for (int idx = tid; idx < NSTATE; idx += QTHREADS) {
        float pr = sr[idx], pi = si[idx];
        float p = pr * pr + pi * pi;
#pragma unroll
        for (int q = 0; q < NQ; ++q) acc[q] += ((idx >> q) & 1) ? -p : p;
    }
#pragma unroll
    for (int off = 16; off > 0; off >>= 1) {
#pragma unroll
        for (int q = 0; q < NQ; ++q) acc[q] += __shfl_down(acc[q], off, 32);
    }
    const int wave = tid >> 5, lane = tid & 31;
    if (lane == 0) {
        for (int q = 0; q < NQ; ++q) red[wave][q] = acc[q];
    }
    __syncthreads();
    if (tid < NQ) {
        float s = 0.f;
        for (int w = 0; w < QTHREADS / 32; ++w) s += red[w][tid];
        qexp[(size_t)bs * NQ + tid] = s;
    }
}

// ---------------------------------------------------------------------------
// Postprocessor MLP with WMMA: q(16x12) -> relu(16) -> relu(8) -> sigmoid(1)
// ---------------------------------------------------------------------------

__global__ __launch_bounds__(32) void postproc_wmma(
    const float* __restrict__ qv,
    const float* __restrict__ P1, const float* __restrict__ c1,
    const float* __restrict__ P2, const float* __restrict__ c2,
    const float* __restrict__ P3, const float* __restrict__ c3,
    float* __restrict__ out)
{
    __shared__ float As[16][32];
    __shared__ float Bs[16][32];
    const int lane = threadIdx.x;
    const int r0   = blockIdx.x * 16;
    const int M    = lane & 15;
    const int half = lane >> 4;
    const int kb8  = half * 8;
    const int N    = lane & 15;
    const int kb16 = half * 16;

    __builtin_prefetch(P2, 0, 0);

    float* Af = &As[0][0];
    float* Bf = &Bs[0][0];
    for (int i = lane; i < 16 * 32; i += 32) { Af[i] = 0.f; Bf[i] = 0.f; }
    __syncthreads();

    // stage q tile (16x12) and P1 (16x12)
    for (int i = lane; i < 16 * NQ; i += 32) {
        As[i / NQ][i % NQ] = qv[(size_t)(r0 + i / NQ) * NQ + i % NQ];
        Bs[i / NQ][i % NQ] = P1[i];
    }
    __syncthreads();

    // ---- Layer 1: relu(q @ P1^T + c1) ----
    v8f acc = tile_matmul(As, Bs, M, N, kb8, kb16);
    float bias = c1[N];
#pragma unroll
    for (int r = 0; r < 8; ++r) As[r + 8 * half][N] = fmaxf(acc[r] + bias, 0.f);
    // stage P2 (8x16) into rows 0..7, zero rows 8..15 (stale P1 data)
    for (int i = lane; i < 8 * 16; i += 32) Bs[i >> 4][i & 15] = P2[i];
    for (int i = lane; i < 8 * 16; i += 32) Bs[8 + (i >> 4)][i & 15] = 0.f;
    __syncthreads();

    // ---- Layer 2: relu(h @ P2^T + c2), N valid < 8 ----
    acc = tile_matmul(As, Bs, M, N, kb8, kb16);
    __syncthreads();
    if (N < 8) {
        bias = c2[N];
#pragma unroll
        for (int r = 0; r < 8; ++r) As[r + 8 * half][N] = fmaxf(acc[r] + bias, 0.f);
    }
    __syncthreads();

    // ---- Layer 3: sigmoid(h @ P3^T + c3), 8 -> 1 dot per row ----
    if (lane < 16) {
        float o = c3[0];
#pragma unroll
        for (int k = 0; k < 8; ++k) o += As[lane][k] * P3[k];
        out[r0 + lane] = 1.f / (1.f + __expf(-o));
    }
}

// ---------------------------------------------------------------------------
extern "C" void kernel_launch(void* const* d_in, const int* in_sizes, int n_in,
                              void* d_out, int out_size, void* d_ws, size_t ws_size,
                              hipStream_t stream)
{
    const float* x  = (const float*)d_in[0];
    const float* W1 = (const float*)d_in[1];
    const float* b1 = (const float*)d_in[2];
    const float* W2 = (const float*)d_in[3];
    const float* b2 = (const float*)d_in[4];
    const float* W3 = (const float*)d_in[5];
    const float* b3 = (const float*)d_in[6];
    const float* qp = (const float*)d_in[7];
    const float* P1 = (const float*)d_in[8];
    const float* c1 = (const float*)d_in[9];
    const float* P2 = (const float*)d_in[10];
    const float* c2 = (const float*)d_in[11];
    const float* P3 = (const float*)d_in[12];
    const float* c3 = (const float*)d_in[13];
    float* out = (float*)d_out;

    float* angles = (float*)d_ws;                 // BATCH*NQ floats
    float* qexp   = angles + (size_t)BATCH * NQ;  // BATCH*NQ floats

    preproc_wmma<<<BATCH / 16, 32, 0, stream>>>(x, W1, b1, W2, b2, W3, b3, angles);
    quantum_sim<<<BATCH, QTHREADS, 0, stream>>>(angles, qp, qexp);
    postproc_wmma<<<BATCH / 16, 32, 0, stream>>>(qexp, P1, c1, P2, c2, P3, c3, out);
}
